// LlamaAttention_45406394253674
// MI455X (gfx1250) — matrix operand order repaired and checked by reading, not compiled, not run on hardware
//
#include <hip/hip_runtime.h>

// ---------------------------------------------------------------------------
// LLaMA attention layer for gfx1250 (CDNA5, wave32, WMMA bf16).
//   L=2048, HID=4096, NH=32, NKV=8, HD=128, MAX=4096
// Pipeline: rmsnorm -> bf16 weight cvt -> WMMA qkv gemms -> rope + cache
//           scatter -> flash-attention (WMMA) -> WMMA out-proj + residual.
// GEMM: 32x64 output tile per wave (B fragments reused across 2 A tiles,
// ~21 flop/byte from L2 vs ~13 for a 16x64 strip).
// ---------------------------------------------------------------------------

#define L_SEQ 2048
#define HID   4096
#define NH    32
#define NKV   8
#define HD    128
#define MAXP  4096
#define GQA   (NH / NKV)

typedef __attribute__((ext_vector_type(16))) __bf16 v16bf;
typedef __attribute__((ext_vector_type(8)))  __bf16 v8bf;
typedef __attribute__((ext_vector_type(8)))  float  v8f;

__device__ __forceinline__ v16bf ld16(const __bf16* p0, const __bf16* p1) {
  v8bf lo = *(const v8bf*)p0;
  v8bf hi = *(const v8bf*)p1;
  return __builtin_shufflevector(lo, hi, 0, 1, 2, 3, 4, 5, 6, 7,
                                         8, 9, 10, 11, 12, 13, 14, 15);
}

__device__ __forceinline__ v8f wmma_bf16(v16bf a, v16bf b, v8f c) {
  // (neg_a, A, neg_b, B, c_mod, C, reuse_a, reuse_b)
  return __builtin_amdgcn_wmma_f32_16x16x32_bf16(false, a, false, b,
                                                 (short)0, c, false, false);
}

// ------------------------------ RMSNorm ------------------------------------
__global__ void __launch_bounds__(256)
rmsnorm_kernel(const float* __restrict__ x, const float* __restrict__ w,
               __bf16* __restrict__ h) {
  int row = blockIdx.x;
  const float* xr = x + (size_t)row * HID;
  __shared__ float red[256];
  float s = 0.f;
  for (int i = threadIdx.x; i < HID; i += 256) { float v = xr[i]; s += v * v; }
  red[threadIdx.x] = s;
  __syncthreads();
  for (int off = 128; off > 0; off >>= 1) {
    if ((int)threadIdx.x < off) red[threadIdx.x] += red[threadIdx.x + off];
    __syncthreads();
  }
  float inv = rsqrtf(red[0] * (1.0f / HID) + 1e-5f);
  for (int i = threadIdx.x; i < HID; i += 256)
    h[(size_t)row * HID + i] = (__bf16)(w[i] * xr[i] * inv);
}

// --------------------------- f32 -> bf16 cvt -------------------------------
__global__ void __launch_bounds__(256)
cvt_bf16(const float* __restrict__ s, __bf16* __restrict__ d, int n) {
  int i = blockIdx.x * 256 + threadIdx.x;
  if (i < n) d[i] = (__bf16)s[i];
}

// ------------------------- WMMA GEMM: C = A*B^T (+R) -----------------------
// A: [M,K] bf16 row-major, B: [N,K] bf16 row-major (weight rows), C: f32.
// One wave computes a 32x64 output tile. M mult of 32, N mult of 64, K of 32.
__global__ void __launch_bounds__(256)
gemm_bf16(const __bf16* __restrict__ A, const __bf16* __restrict__ B,
          const float* __restrict__ R, float* __restrict__ C,
          int M, int N, int K) {
  int wid  = (blockIdx.x * 256 + threadIdx.x) >> 5;
  int lane = threadIdx.x & 31;
  int lo = lane & 15, hi = lane >> 4;
  int ntiles = N >> 6;
  int mt = wid / ntiles, nt = wid - mt * ntiles;
  int m0 = mt << 5, n0 = nt << 6;

  // A operand: lane holds row m0+lo (+16), K chunks at hi*8 and hi*8+16.
  const __bf16* Ap0 = A + (size_t)(m0 + lo) * K + hi * 8;
  const __bf16* Ap1 = Ap0 + (size_t)16 * K;
  // B operand: lane holds column (weight row) n0+t*16+lo, K chunks at hi*8
  // and hi*8+16.
  const __bf16* Bp = B + (size_t)(n0 + lo) * K + hi * 8;
  size_t bs = (size_t)16 * K;

  v8f acc[2][4] = {};
  for (int k = 0; k < K; k += 32) {
    __builtin_prefetch(Bp + k + 1024, 0, 1);
    v16bf a0 = ld16(Ap0 + k, Ap0 + k + 16);
    v16bf a1 = ld16(Ap1 + k, Ap1 + k + 16);
#pragma unroll
    for (int t = 0; t < 4; ++t) {
      v16bf b = ld16(Bp + t * bs + k, Bp + t * bs + k + 16);
      acc[0][t] = wmma_bf16(a0, b, acc[0][t]);
      acc[1][t] = wmma_bf16(a1, b, acc[1][t]);
    }
  }

#pragma unroll
  for (int u = 0; u < 2; ++u) {
#pragma unroll
    for (int t = 0; t < 4; ++t) {
      int col = n0 + t * 16 + lo;
#pragma unroll
      for (int r = 0; r < 8; ++r) {
        int row = m0 + u * 16 + r + (hi << 3);
        size_t idx = (size_t)row * N + col;
        float v = acc[u][t][r];
        C[idx] = R ? (R[idx] + v) : v;
      }
    }
  }
}

// ------------------------------ RoPE (q) -----------------------------------
__global__ void __launch_bounds__(256)
rope_q_kernel(const float* __restrict__ qf, const float* __restrict__ cosT,
              const float* __restrict__ sinT, __bf16* __restrict__ qb) {
  int t = blockIdx.x * 256 + threadIdx.x;       // L*NH*64 work items
  int d = t & 63;
  int head = (t >> 6) & (NH - 1);
  int l = t >> 11;                              // / (NH*64)
  size_t base = ((size_t)l * NH + head) * HD;
  float x0 = qf[base + d], x1 = qf[base + d + 64];
  float c0 = cosT[l * HD + d],      s0 = sinT[l * HD + d];
  float c1 = cosT[l * HD + d + 64], s1 = sinT[l * HD + d + 64];
  qb[base + d]      = (__bf16)(x0 * c0 - x1 * s0);
  qb[base + d + 64] = (__bf16)(x1 * c1 + x0 * s1);
}

// --------------------- RoPE (k) + cache scatter + bf16 copy ----------------
__global__ void __launch_bounds__(256)
rope_k_kernel(const float* __restrict__ kf, const float* __restrict__ cosT,
              const float* __restrict__ sinT, const long long* __restrict__ pos,
              float* __restrict__ kc_out, __bf16* __restrict__ Kbf) {
  int t = blockIdx.x * 256 + threadIdx.x;       // L*NKV*64 work items
  int d = t & 63;
  int kvh = (t >> 6) & (NKV - 1);
  int l = t >> 9;                               // / (NKV*64)
  size_t base = ((size_t)l * NKV + kvh) * HD;
  float x0 = kf[base + d], x1 = kf[base + d + 64];
  float c0 = cosT[l * HD + d],      s0 = sinT[l * HD + d];
  float c1 = cosT[l * HD + d + 64], s1 = sinT[l * HD + d + 64];
  float o0 = x0 * c0 - x1 * s0;
  float o1 = x1 * c1 + x0 * s1;
  long long p = pos[l];
  size_t cb = ((size_t)p * NKV + kvh) * HD;
  kc_out[cb + d] = o0;
  kc_out[cb + d + 64] = o1;
  size_t kb = ((size_t)kvh * L_SEQ + l) * HD;   // [kv][l][hd] bf16 copy
  Kbf[kb + d] = (__bf16)o0;
  Kbf[kb + d + 64] = (__bf16)o1;
}

// ------------------- V cache scatter + transposed bf16 copy ----------------
__global__ void __launch_bounds__(256)
scatter_v_kernel(const float* __restrict__ vf, const long long* __restrict__ pos,
                 float* __restrict__ vc_out, __bf16* __restrict__ Vt) {
  int t = blockIdx.x * 256 + threadIdx.x;       // L*NKV*HD work items
  int d = t & 127;
  int kvh = (t >> 7) & (NKV - 1);
  int l = t >> 10;                              // / (NKV*HD)
  float v = vf[((size_t)l * NKV + kvh) * HD + d];
  long long p = pos[l];
  vc_out[((size_t)p * NKV + kvh) * HD + d] = v;
  Vt[((size_t)kvh * HD + d) * L_SEQ + l] = (__bf16)v;  // [kv][hd][l]
}

// ----------------- Flash attention: one wave per (head, 16-row q tile) -----
__global__ void __launch_bounds__(32)
attn_fa(const __bf16* __restrict__ qb, const __bf16* __restrict__ Kb,
        const __bf16* __restrict__ Vt, __bf16* __restrict__ attn) {
  int h = blockIdx.x;
  int mt = blockIdx.y;
  int lane = threadIdx.x;
  int lo = lane & 15, hi = lane >> 4;
  int kv = h / GQA;
  int l0 = mt << 4;

  __shared__ float pl[16][36];  // P staging (C-layout -> A-layout transpose)

  // Q A-operands: 4 tiles cover HD=128 (K=32 each)
  const __bf16* qp = qb + ((size_t)(l0 + lo) * NH + h) * HD + hi * 16;
  v16bf Aq[4];
#pragma unroll
  for (int j = 0; j < 4; ++j) Aq[j] = ld16(qp + 32 * j, qp + 32 * j + 8);

  v8f acc[8] = {};
  float m[8], lsum[8];
#pragma unroll
  for (int r = 0; r < 8; ++r) { m[r] = -3.0e38f; lsum[r] = 0.f; }

  const float scale = 0.08838834764831845f;  // 1/sqrt(128)
  const __bf16* Kbase = Kb + (size_t)kv * L_SEQ * HD;
  const __bf16* Vbase = Vt + (size_t)kv * HD * L_SEQ;
  int kend = l0 + 15;

  for (int kb = 0; kb <= kend; kb += 32) {
    // ---- S = Q * K^T for 32 key columns (two 16x16 C tiles) ----
    const __bf16* kp0 = Kbase + (size_t)(kb + lo) * HD + hi * 16;
    const __bf16* kp1 = Kbase + (size_t)(kb + 16 + lo) * HD + hi * 16;
    v8f s0 = {}, s1 = {};
#pragma unroll
    for (int j = 0; j < 4; ++j) {
      v16bf b0 = ld16(kp0 + 32 * j, kp0 + 32 * j + 8);
      v16bf b1 = ld16(kp1 + 32 * j, kp1 + 32 * j + 8);
      s0 = wmma_bf16(Aq[j], b0, s0);
      s1 = wmma_bf16(Aq[j], b1, s1);
    }

    // ---- causal mask + online softmax (row across 16-lane half) ----
    float alpha[8];
#pragma unroll
    for (int r = 0; r < 8; ++r) {
      int qpos = l0 + r + (hi << 3);
      float v0 = s0[r] * scale;
      float v1 = s1[r] * scale;
      if (kb + lo > qpos)      v0 = -3.0e38f;
      if (kb + 16 + lo > qpos) v1 = -3.0e38f;
      float mx = fmaxf(v0, v1);
#pragma unroll
      for (int o = 1; o < 16; o <<= 1) mx = fmaxf(mx, __shfl_xor(mx, o, 32));
      float mn = fmaxf(m[r], mx);
      float a  = __expf(m[r] - mn);
      float p0 = __expf(v0 - mn);
      float p1 = __expf(v1 - mn);
      float rs = p0 + p1;
#pragma unroll
      for (int o = 1; o < 16; o <<= 1) rs += __shfl_xor(rs, o, 32);
      m[r] = mn;
      lsum[r] = lsum[r] * a + rs;
      alpha[r] = a;
      int row = r + (hi << 3);
      pl[row][lo] = p0;
      pl[row][16 + lo] = p1;
    }
    __syncthreads();

    // ---- re-layout P into A operand (16x32, K chunks hi*8 / hi*8+16) ----
    v16bf Ap;
    const float* pr = &pl[lo][hi * 8];
#pragma unroll
    for (int i = 0; i < 8; ++i) {
      Ap[i]     = (__bf16)pr[i];
      Ap[8 + i] = (__bf16)pr[16 + i];
    }
    __syncthreads();

    // ---- O = O*alpha + P*V ----
#pragma unroll
    for (int c = 0; c < 8; ++c) {
#pragma unroll
      for (int r = 0; r < 8; ++r) acc[c][r] *= alpha[r];
      const __bf16* vp = Vbase + (size_t)(c * 16 + lo) * L_SEQ + kb + hi * 8;
      v16bf bv = ld16(vp, vp + 16);
      acc[c] = wmma_bf16(Ap, bv, acc[c]);
    }
  }

  // ---- normalize + store bf16 attn output [L, NH*HD] ----
#pragma unroll
  for (int c = 0; c < 8; ++c) {
    int col = h * HD + c * 16 + lo;
#pragma unroll
    for (int r = 0; r < 8; ++r) {
      int row = l0 + r + (hi << 3);
      attn[(size_t)row * (NH * HD) + col] = (__bf16)(acc[c][r] / lsum[r]);
    }
  }
}

// ---------------------------------------------------------------------------
extern "C" void kernel_launch(void* const* d_in, const int* in_sizes, int n_in,
                              void* d_out, int out_size, void* d_ws, size_t ws_size,
                              hipStream_t stream) {
  const float*     x    = (const float*)d_in[0];
  const float*     cosT = (const float*)d_in[1];
  const float*     sinT = (const float*)d_in[2];
  const long long* pos  = (const long long*)d_in[3];
  // d_in[4] = seq_len (compile-time L_SEQ)
  const float*     kc_in = (const float*)d_in[5];
  const float*     vc_in = (const float*)d_in[6];
  const float*     lnw  = (const float*)d_in[7];
  const float*     Wq   = (const float*)d_in[8];
  const float*     Wk   = (const float*)d_in[9];
  const float*     Wv   = (const float*)d_in[10];
  const float*     Wo   = (const float*)d_in[11];

  float* out    = (float*)d_out;
  float* kc_out = out + (size_t)L_SEQ * HID;
  float* vc_out = kc_out + (size_t)MAXP * NKV * HD;

  char* w = (char*)d_ws;
  auto alloc = [&](size_t bytes) {
    char* p = w;
    w += (bytes + 255) & ~(size_t)255;
    return (void*)p;
  };
  __bf16* h_b    = (__bf16*)alloc((size_t)L_SEQ * HID * 2);
  __bf16* Wq_b   = (__bf16*)alloc((size_t)HID * HID * 2);
  __bf16* Wk_b   = (__bf16*)alloc((size_t)NKV * HD * HID * 2);
  __bf16* Wv_b   = (__bf16*)alloc((size_t)NKV * HD * HID * 2);
  __bf16* Wo_b   = (__bf16*)alloc((size_t)HID * HID * 2);
  float*  q_f    = (float*)alloc((size_t)L_SEQ * HID * 4);
  float*  k_f    = (float*)alloc((size_t)L_SEQ * NKV * HD * 4);
  float*  v_f    = (float*)alloc((size_t)L_SEQ * NKV * HD * 4);
  __bf16* q_b    = (__bf16*)alloc((size_t)L_SEQ * HID * 2);
  __bf16* Kbf    = (__bf16*)alloc((size_t)NKV * L_SEQ * HD * 2);
  __bf16* Vtb    = (__bf16*)alloc((size_t)NKV * HD * L_SEQ * 2);
  __bf16* attn_b = (__bf16*)alloc((size_t)L_SEQ * HID * 2);

  // Caches: copy input state, then scatter new rows at position_ids.
  hipMemcpyAsync(kc_out, kc_in, (size_t)MAXP * NKV * HD * 4,
                 hipMemcpyDeviceToDevice, stream);
  hipMemcpyAsync(vc_out, vc_in, (size_t)MAXP * NKV * HD * 4,
                 hipMemcpyDeviceToDevice, stream);

  rmsnorm_kernel<<<L_SEQ, 256, 0, stream>>>(x, lnw, h_b);

  cvt_bf16<<<(HID * HID) / 256, 256, 0, stream>>>(Wq, Wq_b, HID * HID);
  cvt_bf16<<<(NKV * HD * HID) / 256, 256, 0, stream>>>(Wk, Wk_b, NKV * HD * HID);
  cvt_bf16<<<(NKV * HD * HID) / 256, 256, 0, stream>>>(Wv, Wv_b, NKV * HD * HID);
  cvt_bf16<<<(HID * HID) / 256, 256, 0, stream>>>(Wo, Wo_b, HID * HID);

  // 32x64 tiles: waves = (M/32)*(N/64), 8 waves per 256-thread block.
  gemm_bf16<<<(L_SEQ / 32) * (HID / 64) / 8, 256, 0, stream>>>(
      h_b, Wq_b, nullptr, q_f, L_SEQ, HID, HID);
  gemm_bf16<<<(L_SEQ / 32) * ((NKV * HD) / 64) / 8, 256, 0, stream>>>(
      h_b, Wk_b, nullptr, k_f, L_SEQ, NKV * HD, HID);
  gemm_bf16<<<(L_SEQ / 32) * ((NKV * HD) / 64) / 8, 256, 0, stream>>>(
      h_b, Wv_b, nullptr, v_f, L_SEQ, NKV * HD, HID);

  rope_q_kernel<<<(L_SEQ * NH * 64) / 256, 256, 0, stream>>>(q_f, cosT, sinT, q_b);
  rope_k_kernel<<<(L_SEQ * NKV * 64) / 256, 256, 0, stream>>>(
      k_f, cosT, sinT, pos, kc_out, Kbf);
  scatter_v_kernel<<<(L_SEQ * NKV * HD) / 256, 256, 0, stream>>>(
      v_f, pos, vc_out, Vtb);

  attn_fa<<<dim3(NH, L_SEQ / 16), 32, 0, stream>>>(q_b, Kbf, Vtb, attn_b);

  gemm_bf16<<<(L_SEQ / 32) * (HID / 64) / 8, 256, 0, stream>>>(
      attn_b, Wo_b, x, out, L_SEQ, HID, HID);
}
